// DSGPM_61967788147234
// MI455X (gfx1250) — matrix operand order repaired and checked by reading, not compiled, not run on hardware
//
#include <hip/hip_runtime.h>
#include <hip/hip_bf16.h>

#define HID     64
#define NNODES  16384
#define NEDGES  65536

typedef __attribute__((ext_vector_type(16))) _Float16 v16h;
typedef __attribute__((ext_vector_type(8)))  _Float16 v8h;
typedef __attribute__((ext_vector_type(2)))  _Float16 v2h;
typedef __attribute__((ext_vector_type(8)))  float    v8f;

#define MODE_EW    0
#define MODE_G1    1
#define MODE_G2    2
#define MODE_RELUH 3
#define MODE_F     4

// ---------------------------------------------------------------------------
// Pack a K x N f32 matrix (optionally given as N x K, transpose=1) into the
// WMMA-B lane layout for 16x16x32 f16:
//   dst[((nt*KC + kc)*32 + lane)*16 + j] = B[kc*32 + (lane>>4)*16 + j][nt*16 + (lane&15)]
// so each lane's 16 halves (32 B) are contiguous in memory.
// ---------------------------------------------------------------------------
__global__ void k_pack_b(_Float16* __restrict__ dst, const float* __restrict__ src,
                         int K, int N, int transpose) {
  int idx = blockIdx.x * blockDim.x + threadIdx.x;
  int total = (N >> 4) * (K >> 5) * 512;
  if (idx >= total) return;
  int j    = idx & 15;
  int lane = (idx >> 4) & 31;
  int rest = idx >> 9;
  int kcN  = K >> 5;
  int kc   = rest % kcN;
  int nt   = rest / kcN;
  int n = (nt << 4) + (lane & 15);
  int k = (kc << 5) + ((lane >> 4) << 4) + j;
  float v = transpose ? src[(size_t)n * K + k] : src[(size_t)k * N + n];
  dst[idx] = (_Float16)v;
}

// s = relu(emb_table[x]); keep f32 state + f16 copy (WMMA A operand)
__global__ void k_embed(const int* __restrict__ x, const float* __restrict__ emb,
                        float* __restrict__ s, _Float16* __restrict__ sh) {
  int idx = blockIdx.x * blockDim.x + threadIdx.x;
  if (idx >= NNODES * HID) return;
  int n = idx >> 6, c = idx & 63;
  float v = emb[(size_t)x[n] * HID + c];
  v = v > 0.f ? v : 0.f;
  s[idx]  = v;
  sh[idx] = (_Float16)v;
}

// r[e,k] = relu(edge_attr[e,:4] @ ew1[:,k] + eb1[k])  (f16, A operand of ew GEMM)
__global__ void k_edge_hidden(const float* __restrict__ ea, const float* __restrict__ W1,
                              const float* __restrict__ b1, _Float16* __restrict__ r) {
  int idx = blockIdx.x * blockDim.x + threadIdx.x;
  if (idx >= NEDGES * 128) return;
  int e = idx >> 7, k = idx & 127;
  float v = b1[k];
#pragma unroll
  for (int j = 0; j < 4; ++j) v += ea[e * 4 + j] * W1[j * 128 + k];
  r[idx] = (_Float16)(v > 0.f ? v : 0.f);
}

__global__ void k_zero(float* __restrict__ p, int n) {
  int i = blockIdx.x * blockDim.x + threadIdx.x;
  if (i < n) p[i] = 0.f;
}

// ---------------------------------------------------------------------------
// Generic WMMA f16 GEMM: C(MxN) = A(MxK, row-major f16) x Bp(packed f16) with
// a mode-selected epilogue. One wave computes one 16x16 tile; K in chunks of 32
// via v_wmma_f32_16x16x32_f16. C layout per CDNA5 ISA 7.12.2; A/B use contiguous k = 16h + s.
// ---------------------------------------------------------------------------
__global__ void __launch_bounds__(256) k_gemm(
    const _Float16* __restrict__ A, const _Float16* __restrict__ Bp,
    int M, int N, int K, int mode,
    const float* __restrict__ bias, const float* __restrict__ bias2,
    const float* __restrict__ aux,
    _Float16* __restrict__ outH, float* __restrict__ outF) {
  int lane   = threadIdx.x & 31;
  int wave   = blockIdx.x * (blockDim.x >> 5) + (threadIdx.x >> 5);
  int ntiles = N >> 4;
  int kcN    = K >> 5;
  long total = (long)(M >> 4) * ntiles;
  if (wave >= total) return;
  int nt = wave % ntiles;
  int mt = wave / ntiles;

  // A loader: lane L holds row mt*16 + (L&15); K-offsets {kb..kb+15}
  int arow = (mt << 4) + (lane & 15);
  int kb   = (lane >> 4) << 4;                    // 0 or 16
  const _Float16* arow_p = A + (size_t)arow * K;
  const _Float16* bp = Bp + (((size_t)nt * kcN) << 9) + ((size_t)lane << 4);

  v8f acc = {};
  for (int kc = 0; kc < kcN; ++kc) {
    const _Float16* ap = arow_p + (kc << 5) + kb;
    v8h lo = *(const v8h*)ap;                     // K = kb+0 .. kb+7
    v8h hi = *(const v8h*)(ap + 8);               // K = kb+8 .. kb+15
    v16h a;
#pragma unroll
    for (int i = 0; i < 8; ++i) { a[i] = lo[i]; a[8 + i] = hi[i]; }
    v16h b = *(const v16h*)(bp + ((size_t)kc << 9));
    acc = __builtin_amdgcn_wmma_f32_16x16x32_f16(false, a, false, b,
                                                 (short)0, acc, false, false);
  }

  // C layout: VGPR r -> row = mt*16 + (lane>>4)*8 + r, col = nt*16 + (lane&15)
  int col   = (nt << 4) + (lane & 15);
  int rbase = (lane >> 4) << 3;
#pragma unroll
  for (int rr = 0; rr < 8; ++rr) {
    int row = (mt << 4) + rbase + rr;
    float v = acc[rr];
    if (mode == MODE_EW) {                        // ew = r@W2 + b2 -> f16
      outH[(size_t)row * N + col] = (_Float16)(v + bias[col]);
    } else if (mode == MODE_G1) {                 // [root_w | w_hh^T] fused
      if (col < HID) {                            // m = relu(agg + s@root_w + conv_b)
        float m = v + bias[col] + aux[(size_t)row * HID + col];
        m = m > 0.f ? m : 0.f;
        outH[(size_t)row * HID + col] = (_Float16)m;
      } else {                                    // gh = s@w_hh^T + b_hh
        outF[(size_t)row * 192 + (col - HID)] = v + bias2[col - HID];
      }
    } else if (mode == MODE_G2) {                 // gi = m@w_ih^T + b_ih
      outF[(size_t)row * 192 + col] = v + bias[col];
    } else if (mode == MODE_RELUH) {              // o1 = relu(s@ow1 + ob1) -> f16
      float m = v + bias[col];
      outH[(size_t)row * N + col] = (_Float16)(m > 0.f ? m : 0.f);
    } else {                                      // o2 = o1@ow2 + ob2 -> f32
      outF[(size_t)row * N + col] = v + bias[col];
    }
  }
}

// ---------------------------------------------------------------------------
// msg[e] = s[src[e]] @ ew[e]  (1x64 * 64x64, f16 matrix), scatter-add to agg.
// One wave per edge; lane l produces outputs 2l, 2l+1; bandwidth-bound on ew.
// ---------------------------------------------------------------------------
__global__ void __launch_bounds__(256) k_msg(
    const float* __restrict__ s, const _Float16* __restrict__ ew,
    const int* __restrict__ src, const int* __restrict__ dst,
    float* __restrict__ agg, int E) {
  int lane   = threadIdx.x & 31;
  int wave   = blockIdx.x * (blockDim.x >> 5) + (threadIdx.x >> 5);
  int nwaves = gridDim.x * (blockDim.x >> 5);
  for (int e = wave; e < E; e += nwaves) {
    if (e + nwaves < E)                           // stream next edge's 8KB matrix
      __builtin_prefetch(ew + (size_t)(e + nwaves) * 4096 + lane * 128, 0, 1);
    int sn = src[e], dn = dst[e];
    float x0 = s[(size_t)sn * HID + lane];
    float x1 = s[(size_t)sn * HID + 32 + lane];
    const _Float16* W = ew + (size_t)e * 4096;
    float a0 = 0.f, a1 = 0.f;
#pragma unroll 8
    for (int h = 0; h < 32; ++h) {
      float xh = __shfl(x0, h, 32);
      v2h w = *(const v2h*)(W + h * HID + 2 * lane);
      a0 += xh * (float)w.x;
      a1 += xh * (float)w.y;
    }
#pragma unroll 8
    for (int h = 0; h < 32; ++h) {
      float xh = __shfl(x1, h, 32);
      v2h w = *(const v2h*)(W + (h + 32) * HID + 2 * lane);
      a0 += xh * (float)w.x;
      a1 += xh * (float)w.y;
    }
    atomicAdd(&agg[(size_t)dn * HID + 2 * lane], a0);
    atomicAdd(&agg[(size_t)dn * HID + 2 * lane + 1], a1);
  }
}

// GRU cell elementwise: s_new = (1-z)*n + z*s
__global__ void k_gru(const float* __restrict__ gi, const float* __restrict__ gh,
                      float* __restrict__ s, _Float16* __restrict__ sh) {
  int idx = blockIdx.x * blockDim.x + threadIdx.x;
  if (idx >= NNODES * HID) return;
  int n = idx >> 6, c = idx & 63;
  float ir = gi[(size_t)n * 192 + c];
  float iz = gi[(size_t)n * 192 + 64 + c];
  float in = gi[(size_t)n * 192 + 128 + c];
  float hr = gh[(size_t)n * 192 + c];
  float hz = gh[(size_t)n * 192 + 64 + c];
  float hn = gh[(size_t)n * 192 + 128 + c];
  float r = 1.f / (1.f + __expf(-(ir + hr)));
  float z = 1.f / (1.f + __expf(-(iz + hz)));
  float nn = tanhf(in + r * hn);
  float hnew = (1.f - z) * nn + z * s[idx];
  s[idx]  = hnew;
  sh[idx] = (_Float16)hnew;
}

// feat = [o2, one_hot(x)] / max(||.||, eps); one wave per node.
__global__ void __launch_bounds__(256) k_final(const float* __restrict__ o2,
                                               const int* __restrict__ x,
                                               float* __restrict__ out) {
  int lane = threadIdx.x & 31;
  int node = blockIdx.x * (blockDim.x >> 5) + (threadIdx.x >> 5);
  if (node >= NNODES) return;
  float v0 = o2[(size_t)node * HID + lane];
  float v1 = o2[(size_t)node * HID + 32 + lane];
  float ss = v0 * v0 + v1 * v1;
#pragma unroll
  for (int off = 16; off > 0; off >>= 1) ss += __shfl_xor(ss, off, 32);
  ss += 1.0f;                                     // one-hot contributes exactly 1
  float inv = 1.f / fmaxf(sqrtf(ss), 1e-12f);
  out[(size_t)node * 84 + lane]      = v0 * inv;
  out[(size_t)node * 84 + 32 + lane] = v1 * inv;
  if (lane < 20) out[(size_t)node * 84 + 64 + lane] = (lane == x[node]) ? inv : 0.f;
}

extern "C" void kernel_launch(void* const* d_in, const int* in_sizes, int n_in,
                              void* d_out, int out_size, void* d_ws, size_t ws_size,
                              hipStream_t stream) {
  (void)in_sizes; (void)n_in; (void)out_size; (void)ws_size;
  const int*   x      = (const int*)d_in[0];
  const int*   eidx   = (const int*)d_in[1];
  const float* ea     = (const float*)d_in[2];
  const float* emb    = (const float*)d_in[3];
  const float* ew1    = (const float*)d_in[4];
  const float* eb1    = (const float*)d_in[5];
  const float* ew2    = (const float*)d_in[6];
  const float* eb2    = (const float*)d_in[7];
  const float* root_w = (const float*)d_in[8];
  const float* conv_b = (const float*)d_in[9];
  const float* w_ih   = (const float*)d_in[10];
  const float* w_hh   = (const float*)d_in[11];
  const float* b_ih   = (const float*)d_in[12];
  const float* b_hh   = (const float*)d_in[13];
  const float* ow1    = (const float*)d_in[14];
  const float* ob1    = (const float*)d_in[15];
  const float* ow2    = (const float*)d_in[16];
  const float* ob2    = (const float*)d_in[17];
  float* out = (float*)d_out;
  const int* srcp = eidx;
  const int* dstp = eidx + NEDGES;

  // Workspace carve (dominant: ew in f16 = 512 MiB to halve HBM traffic)
  uint8_t* p = (uint8_t*)d_ws;
  auto carve = [&](size_t bytes) -> void* {
    void* q = (void*)p; p += (bytes + 255) & ~(size_t)255; return q;
  };
  _Float16* r_h  = (_Float16*)carve((size_t)NEDGES * 128 * 2);
  _Float16* ewm  = (_Float16*)carve((size_t)NEDGES * 4096 * 2);
  _Float16* W2p  = (_Float16*)carve((size_t)4096 * 128 * 2);
  _Float16* Wc1p = (_Float16*)carve((size_t)256 * 64 * 2);   // [root_w | w_hh^T]
  _Float16* Wihp = (_Float16*)carve((size_t)192 * 64 * 2);
  _Float16* Wo1p = (_Float16*)carve((size_t)64 * 64 * 2);
  _Float16* Wo2p = (_Float16*)carve((size_t)64 * 64 * 2);
  float*    s    = (float*)   carve((size_t)NNODES * 64 * 4);
  _Float16* s_h  = (_Float16*)carve((size_t)NNODES * 64 * 2);
  float*    agg  = (float*)   carve((size_t)NNODES * 64 * 4);
  _Float16* m_h  = (_Float16*)carve((size_t)NNODES * 64 * 2);
  float*    gh   = (float*)   carve((size_t)NNODES * 192 * 4);
  float*    gi   = (float*)   carve((size_t)NNODES * 192 * 4);
  _Float16* o1_h = (_Float16*)carve((size_t)NNODES * 64 * 2);
  float*    o2   = (float*)   carve((size_t)NNODES * 64 * 4);

  auto packElems = [](int K, int N) { return (N >> 4) * (K >> 5) * 512; };

  // Pack all B operands into WMMA lane layout (f16)
  k_pack_b<<<(packElems(128, 4096) + 255) / 256, 256, 0, stream>>>(W2p, ew2, 128, 4096, 0);
  k_pack_b<<<(packElems(64, 64) + 255) / 256, 256, 0, stream>>>(Wc1p, root_w, 64, 64, 0);
  k_pack_b<<<(packElems(64, 192) + 255) / 256, 256, 0, stream>>>(Wc1p + 4096, w_hh, 64, 192, 1);
  k_pack_b<<<(packElems(64, 192) + 255) / 256, 256, 0, stream>>>(Wihp, w_ih, 64, 192, 1);
  k_pack_b<<<(packElems(64, 64) + 255) / 256, 256, 0, stream>>>(Wo1p, ow1, 64, 64, 0);
  k_pack_b<<<(packElems(64, 64) + 255) / 256, 256, 0, stream>>>(Wo2p, ow2, 64, 64, 0);

  // s = relu(emb[x]);  r = relu(ea@W1 + b1)
  k_embed<<<(NNODES * 64 + 255) / 256, 256, 0, stream>>>(x, emb, s, s_h);
  k_edge_hidden<<<(NEDGES * 128 + 255) / 256, 256, 0, stream>>>(ea, ew1, eb1, r_h);

  // ew = r @ W2 + b2   (M=65536, N=4096, K=128) -> WMMA, f16 output
  {
    long waves = (long)(NEDGES / 16) * (4096 / 16);
    k_gemm<<<(int)((waves + 7) / 8), 256, 0, stream>>>(
        r_h, W2p, NEDGES, 4096, 128, MODE_EW, eb2, nullptr, nullptr, ewm, nullptr);
  }

  for (int it = 0; it < 3; ++it) {
    k_zero<<<(NNODES * 64 + 255) / 256, 256, 0, stream>>>(agg, NNODES * 64);
    k_msg<<<NEDGES / 32, 256, 0, stream>>>(s, ewm, srcp, dstp, agg, NEDGES);
    // fused: m = relu(agg + s@root_w + conv_b);  gh = s@w_hh^T + b_hh
    long w1 = (long)(NNODES / 16) * (256 / 16);
    k_gemm<<<(int)((w1 + 7) / 8), 256, 0, stream>>>(
        s_h, Wc1p, NNODES, 256, 64, MODE_G1, conv_b, b_hh, agg, m_h, gh);
    // gi = m@w_ih^T + b_ih
    long w2 = (long)(NNODES / 16) * (192 / 16);
    k_gemm<<<(int)((w2 + 7) / 8), 256, 0, stream>>>(
        m_h, Wihp, NNODES, 192, 64, MODE_G2, b_ih, nullptr, nullptr, nullptr, gi);
    k_gru<<<(NNODES * 64 + 255) / 256, 256, 0, stream>>>(gi, gh, s, s_h);
  }

  // output MLP + L2 normalize with one-hot
  long wo = (long)(NNODES / 16) * (64 / 16);
  k_gemm<<<(int)((wo + 7) / 8), 256, 0, stream>>>(
      s_h, Wo1p, NNODES, 64, 64, MODE_RELUH, ob1, nullptr, nullptr, o1_h, nullptr);
  k_gemm<<<(int)((wo + 7) / 8), 256, 0, stream>>>(
      o1_h, Wo2p, NNODES, 64, 64, MODE_F, ob2, nullptr, nullptr, nullptr, o2);
  k_final<<<NNODES / 8, 256, 0, stream>>>(o2, x, out);
}
